// BidirRWKV6TemperatureTimeMix_13924283974067
// MI455X (gfx1250) — compile-verified, matrix operand order checked by reading, not run on hardware
//
#include <hip/hip_runtime.h>
#include <hip/hip_bf16.h>
#include <math.h>

typedef __attribute__((ext_vector_type(16))) __bf16 v16bf;
typedef __attribute__((ext_vector_type(8)))  __bf16 v8bf;
typedef __attribute__((ext_vector_type(8)))  float  v8f;

// gcc-style int4 to exactly match the async-LDS builtin's pointer params
typedef int v4i_gcc __attribute__((__vector_size__(16)));
typedef __attribute__((address_space(1))) v4i_gcc gv4i;   // global
typedef __attribute__((address_space(3))) v4i_gcc lv4i;   // LDS

#define T_ 2048
#define D_ 1024
#define H_ 16
#define KH_ 64
#define MIDT 1024   // T/2

// ---------------------------------------------------------------------------
// CDNA5 async global->LDS copies (ASYNCcnt), guarded so the file still
// compiles on toolchains without the builtins.
// ---------------------------------------------------------------------------
#if __has_builtin(__builtin_amdgcn_global_load_async_to_lds_b128)
#define HAVE_ASYNC_LDS 1
#else
#define HAVE_ASYNC_LDS 0
#endif

#if HAVE_ASYNC_LDS
#if __has_builtin(__builtin_amdgcn_s_wait_asynccnt)
#define WAIT_ASYNC(n) __builtin_amdgcn_s_wait_asynccnt(n)
#else
#define WAIT_ASYNC(n) asm volatile("s_wait_asynccnt %0" ::"i"(n) : "memory")
#endif
#else
#define WAIT_ASYNC(n) ((void)0)
#endif

// ---------------------------------------------------------------------------
// WMMA helpers (gfx1250, wave32): V_WMMA_F32_16X16X32_BF16
// ---------------------------------------------------------------------------
__device__ __forceinline__ v8f wmma_bf16(v16bf a, v16bf b, v8f c) {
  return __builtin_amdgcn_wmma_f32_16x16x32_bf16(false, a, false, b, (short)0, c,
                                                 false, false);
}

// 16x32 bf16 operand fragment from LDS, K contiguous per row (stride ld):
//   lanes 0-15 : row M = lane,    K = {0..7} in v0-3, {16..23} in v4-7
//   lanes 16-31: row M = lane-16, K = {8..15},        {24..31}
__device__ __forceinline__ v16bf frag_ld(const __bf16* base, int ld, int lane) {
  int r  = lane & 15;
  int kh = lane >> 4;
  const __bf16* p = base + r * ld + kh * 8;
  v8bf lo = *(const v8bf*)(p);
  v8bf hi = *(const v8bf*)(p + 16);
  v16bf o;
#pragma unroll
  for (int i = 0; i < 8; ++i) { o[i] = lo[i]; o[8 + i] = hi[i]; }
  return o;
}

__device__ __forceinline__ float clip60(float v) {
  return fminf(fmaxf(v, -60.f), 60.f);
}

// Stage a 128x32 bf16 tile (rows r0.., cols k0..k0+31 of a row-major matrix
// with row stride ld) into LDS laid out [128][32].  2 x 16B chunks per thread.
__device__ __forceinline__ void stage_tile(const __bf16* __restrict__ g, int ld,
                                           int r0, int k0,
                                           __bf16* __restrict__ l, int tid) {
#pragma unroll
  for (int it = 0; it < 2; ++it) {
    int q = tid + (it << 8);
    int r = q >> 2, c8 = (q & 3) << 3;
    const __bf16* gp = g + (size_t)(r0 + r) * ld + k0 + c8;
    __bf16* lp = l + r * 32 + c8;
#if HAVE_ASYNC_LDS
    __builtin_amdgcn_global_load_async_to_lds_b128(
        (gv4i*)(__attribute__((address_space(1))) void*)(void*)gp,
        (lv4i*)(__attribute__((address_space(3))) void*)lp, 0, 0);
#else
    *(uint4*)lp = *(const uint4*)gp;
#endif
  }
}

// ---------------------------------------------------------------------------
// Weight prep: B[K][N] f32 -> Bt[Npad][K] bf16 (transposed, zero padded rows)
// ---------------------------------------------------------------------------
__global__ __launch_bounds__(256) void cvt_transpose_k(
    const float* __restrict__ B, __bf16* __restrict__ Bt, int N, int K,
    int Npad) {
  int i = blockIdx.x * 256 + threadIdx.x;
  if (i >= Npad * K) return;
  int n = i / K, k = i - n * K;
  Bt[i] = (n < N) ? (__bf16)B[(size_t)k * N + n] : (__bf16)0.f;
}

// ---------------------------------------------------------------------------
// Token shift + mixing projection: xxx = tanh((x + dx*maa_x) @ maa_w1)
// ---------------------------------------------------------------------------
__global__ __launch_bounds__(256) void prep_k(const float* __restrict__ x,
                                              const float* __restrict__ maa_x,
                                              const float* __restrict__ maa_w1,
                                              float* __restrict__ xxx) {
  int t = blockIdx.x;
  __shared__ float xm[D_];
#pragma unroll
  for (int it = 0; it < 4; ++it) {
    int d = threadIdx.x + it * 256;
    float xc   = x[(size_t)t * D_ + d];
    float prev = (t > 0)      ? x[(size_t)(t - 1) * D_ + d] : 0.f;
    float nxt  = (t < T_ - 1) ? x[(size_t)(t + 1) * D_ + d] : 0.f;
    float dx   = 0.5f * (prev + nxt) - xc;
    xm[d] = xc + dx * maa_x[d];
  }
  __syncthreads();
  for (int j = threadIdx.x; j < 160; j += 256) {
    float s = 0.f;
    for (int d = 0; d < D_; ++d) s += xm[d] * maa_w1[(size_t)d * 160 + j];
    xxx[(size_t)t * 160 + j] = tanhf(s);
  }
}

// ---------------------------------------------------------------------------
// m = einsum('btfm,fmd->fbtd', xxx, maa_w2); x? = x + dx*(maa_? + m_?)  (bf16)
// ---------------------------------------------------------------------------
__global__ __launch_bounds__(256) void mix_k(
    const float* __restrict__ x, const float* __restrict__ xxx,
    const float* __restrict__ maa_w2, const float* __restrict__ maa_w,
    const float* __restrict__ maa_k, const float* __restrict__ maa_v,
    const float* __restrict__ maa_r, const float* __restrict__ maa_g,
    __bf16* __restrict__ xw, __bf16* __restrict__ xk, __bf16* __restrict__ xv,
    __bf16* __restrict__ xr, __bf16* __restrict__ xg) {
  int t = blockIdx.x;
  __shared__ float xl[160];
  for (int j = threadIdx.x; j < 160; j += 256) xl[j] = xxx[(size_t)t * 160 + j];
  __syncthreads();
#pragma unroll
  for (int it = 0; it < 4; ++it) {
    int d = threadIdx.x + it * 256;
    float xc   = x[(size_t)t * D_ + d];
    float prev = (t > 0)      ? x[(size_t)(t - 1) * D_ + d] : 0.f;
    float nxt  = (t < T_ - 1) ? x[(size_t)(t + 1) * D_ + d] : 0.f;
    float dx   = 0.5f * (prev + nxt) - xc;
    float m[5];
#pragma unroll
    for (int f = 0; f < 5; ++f) {
      float s = 0.f;
#pragma unroll
      for (int mm = 0; mm < 32; ++mm)
        s += xl[f * 32 + mm] * maa_w2[(size_t)(f * 32 + mm) * D_ + d];
      m[f] = s;
    }
    size_t idx = (size_t)t * D_ + d;
    xw[idx] = (__bf16)(xc + dx * (maa_w[d] + m[0]));
    xk[idx] = (__bf16)(xc + dx * (maa_k[d] + m[1]));
    xv[idx] = (__bf16)(xc + dx * (maa_v[d] + m[2]));
    xr[idx] = (__bf16)(xc + dx * (maa_r[d] + m[3]));
    xg[idx] = (__bf16)(xc + dx * (maa_g[d] + m[4]));
  }
}

// ---------------------------------------------------------------------------
// GEMM: C[M,N] f32 = A[M,K] bf16 @ Bt[N,K]^T bf16 (Bt pre-transposed).
// 128x128 tile, 8 waves, K-step 32, async double-buffered LDS staging.
// ---------------------------------------------------------------------------
__global__ __launch_bounds__(256) void gemm_bf16_k(
    const __bf16* __restrict__ A, const __bf16* __restrict__ Bt,
    float* __restrict__ C, int N, int Kd) {
  __shared__ __align__(16) __bf16 As[2][128 * 32];
  __shared__ __align__(16) __bf16 Bs[2][128 * 32];
  int tid = threadIdx.x, lane = tid & 31, wave = tid >> 5;
  int row0 = blockIdx.y * 128, col0 = blockIdx.x * 128;
  int wr = wave >> 1, wc = wave & 1;
  v8f acc[2][4];
#pragma unroll
  for (int a = 0; a < 2; ++a)
#pragma unroll
    for (int b = 0; b < 4; ++b)
#pragma unroll
      for (int i = 0; i < 8; ++i) acc[a][b][i] = 0.f;

  int nk = Kd >> 5;
  stage_tile(A, Kd, row0, 0, As[0], tid);
  stage_tile(Bt, Kd, col0, 0, Bs[0], tid);

  for (int i = 0; i < nk; ++i) {
    int buf = i & 1;
    if (i + 1 < nk) {
      stage_tile(A, Kd, row0, (i + 1) << 5, As[buf ^ 1], tid);
      stage_tile(Bt, Kd, col0, (i + 1) << 5, Bs[buf ^ 1], tid);
      WAIT_ASYNC(4);   // tile i (4 oldest ops) landed; tile i+1 in flight
    } else {
      WAIT_ASYNC(0);
    }
    __syncthreads();
#pragma unroll
    for (int mi = 0; mi < 2; ++mi) {
      v16bf a = frag_ld(&As[buf][(wr * 32 + mi * 16) * 32], 32, lane);
#pragma unroll
      for (int ni = 0; ni < 4; ++ni) {
        v16bf b = frag_ld(&Bs[buf][(wc * 64 + ni * 16) * 32], 32, lane);
        acc[mi][ni] = wmma_bf16(a, b, acc[mi][ni]);
      }
    }
    __syncthreads();
  }
  int lm = (lane >> 4) << 3, ln = lane & 15;
#pragma unroll
  for (int mi = 0; mi < 2; ++mi)
#pragma unroll
    for (int ni = 0; ni < 4; ++ni)
#pragma unroll
      for (int i = 0; i < 8; ++i) {
        int gr = row0 + wr * 32 + mi * 16 + lm + i;
        int gc = col0 + wc * 64 + ni * 16 + ln;
        if (gc < N) C[(size_t)gr * N + gc] = acc[mi][ni][i];
      }
}

// ---------------------------------------------------------------------------
// silu in-place
// ---------------------------------------------------------------------------
__global__ __launch_bounds__(256) void silu_k(float* g, int n) {
  int i = blockIdx.x * 256 + threadIdx.x;
  if (i < n) { float v = g[i]; g[i] = v / (1.f + expf(-v)); }
}

// ---------------------------------------------------------------------------
// w_h = -exp(time_decay + tanh(h1) @ decay_w2)
// ---------------------------------------------------------------------------
__global__ __launch_bounds__(256) void wdecay_k(const float* __restrict__ h1,
                                                const float* __restrict__ dw2,
                                                const float* __restrict__ td,
                                                float* __restrict__ wh) {
  int t = blockIdx.x;
  __shared__ float hl[64];
  if (threadIdx.x < 64) hl[threadIdx.x] = tanhf(h1[(size_t)t * 64 + threadIdx.x]);
  __syncthreads();
#pragma unroll
  for (int it = 0; it < 4; ++it) {
    int d = threadIdx.x + it * 256;
    float s = td[d];
#pragma unroll
    for (int j = 0; j < 64; ++j) s += hl[j] * dw2[(size_t)j * D_ + d];
    wh[(size_t)t * D_ + d] = -expf(s);
  }
}

// ---------------------------------------------------------------------------
// inclusive cumsum over time of w_h (one thread per channel, coalesced)
// ---------------------------------------------------------------------------
__global__ __launch_bounds__(256) void cumsum_k(const float* __restrict__ wh,
                                                float* __restrict__ cs) {
  int d = blockIdx.x * 256 + threadIdx.x;
  float acc = 0.f;
  for (int t = 0; t < T_; ++t) {
    acc += wh[(size_t)t * D_ + d];
    cs[(size_t)t * D_ + d] = acc;
  }
}

// ---------------------------------------------------------------------------
// Bidirectional decay attention + temperature sharpening + GroupNorm + gate.
// One block = 64 t-rows x 1 head.  Streams 64-wide s-tiles, WMMA bf16.
// ---------------------------------------------------------------------------
__global__ __launch_bounds__(256) void attn_k(
    const float* __restrict__ rB, const float* __restrict__ kB,
    const float* __restrict__ vB, const float* __restrict__ cs,
    const float* __restrict__ wh, const float* __restrict__ log_tau,
    const float* __restrict__ ln_g, const float* __restrict__ ln_b,
    const float* __restrict__ gB, __bf16* __restrict__ yg) {
  __shared__ __align__(16) __bf16 rf[64 * 64];
  __shared__ __align__(16) __bf16 rb[64 * 64];
  __shared__ __align__(16) __bf16 kf[64 * 64];   // reused: A_sharp (bf16)
  __shared__ __align__(16) __bf16 kb[64 * 64];   // reused: v transposed
  __shared__ __align__(16) float  Asm[64 * 64];  // f32 A tile / final Y tile
  __shared__ float rs_o[64], rs_s[64], cmf[64], cmb[64];

  int tid = threadIdx.x, lane = tid & 31, wave = tid >> 5;
  int t0 = blockIdx.x * 64;
  int hd = blockIdx.y;
  int dbase = hd * KH_;
  float tau = expf(log_tau[hd]);

  if (tid < 64) {
    float c = cs[(size_t)MIDT * D_ + dbase + tid];
    cmf[tid] = c;
    cmb[tid] = c - wh[(size_t)MIDT * D_ + dbase + tid];
    rs_o[tid] = 0.f;
    rs_s[tid] = 0.f;
  }
  __syncthreads();

  // r scaled by forward / backward decay factors (bf16, K contiguous)
#pragma unroll
  for (int it = 0; it < 16; ++it) {
    int q = tid + it * 256;
    int tt = q >> 6, c = q & 63;
    int t = t0 + tt, d = dbase + c;
    float csv = cs[(size_t)t * D_ + d];
    float csf = clip60(csv - cmf[c]);
    float csb = clip60(csv - wh[(size_t)t * D_ + d] - cmb[c]);
    float rv = rB[(size_t)t * D_ + d];
    rf[q] = (__bf16)(rv * expf(csf));
    rb[q] = (__bf16)(rv * expf(-csb));
  }

  int mi = wave >> 1;            // wave's 16-row band (0..3)
  int nj = (wave & 1) * 2;       // wave's first 16-col tile (0 or 2)
  int lm = (lane >> 4) << 3, ln = lane & 15;
  v8f yacc[2];
#pragma unroll
  for (int j = 0; j < 2; ++j)
#pragma unroll
    for (int i = 0; i < 8; ++i) yacc[j][i] = 0.f;

  for (int s0 = 0; s0 < T_; s0 += 64) {
    // phase 1: k scaled tiles
#pragma unroll
    for (int it = 0; it < 16; ++it) {
      int q = tid + it * 256;
      int tt = q >> 6, c = q & 63;
      int s = s0 + tt, d = dbase + c;
      float csv = cs[(size_t)s * D_ + d];
      float csf = clip60(csv - cmf[c]);
      float csb = clip60(csv - wh[(size_t)s * D_ + d] - cmb[c]);
      float kv = kB[(size_t)s * D_ + d];
      kf[q] = (__bf16)(kv * expf(-csf));
      kb[q] = (__bf16)(kv * expf(csb));
    }
    __syncthreads();

    // phase 2: A_fwd / A_bwd via WMMA, mask+relu, write f32 tile
    {
      v16bf af0 = frag_ld(&rf[(mi * 16) * 64 + 0], 64, lane);
      v16bf af1 = frag_ld(&rf[(mi * 16) * 64 + 32], 64, lane);
      v16bf ab0 = frag_ld(&rb[(mi * 16) * 64 + 0], 64, lane);
      v16bf ab1 = frag_ld(&rb[(mi * 16) * 64 + 32], 64, lane);
#pragma unroll
      for (int j = 0; j < 2; ++j) {
        int n = nj + j;
        v8f accf, accb;
#pragma unroll
        for (int i = 0; i < 8; ++i) { accf[i] = 0.f; accb[i] = 0.f; }
        v16bf bf0 = frag_ld(&kf[(n * 16) * 64 + 0], 64, lane);
        v16bf bf1 = frag_ld(&kf[(n * 16) * 64 + 32], 64, lane);
        accf = wmma_bf16(af0, bf0, accf);
        accf = wmma_bf16(af1, bf1, accf);
        v16bf bb0 = frag_ld(&kb[(n * 16) * 64 + 0], 64, lane);
        v16bf bb1 = frag_ld(&kb[(n * 16) * 64 + 32], 64, lane);
        accb = wmma_bf16(ab0, bb0, accb);
        accb = wmma_bf16(ab1, bb1, accb);
#pragma unroll
        for (int i = 0; i < 8; ++i) {
          int rloc = mi * 16 + lm + i;
          int gr = t0 + rloc, gc = s0 + n * 16 + ln;
          float aval = (gr >= gc) ? accf[i] : accb[i];
          Asm[rloc * 64 + n * 16 + ln] = fmaxf(aval, 0.f);
        }
      }
    }
    __syncthreads();

    // phase 3: pow(tau) sharpening, row sums, A_sharp->bf16 (into kf),
    //          v tile transposed into kb.
    {
      int rr = tid >> 2, cq = (tid & 3) << 4;
      float so = 0.f, ss = 0.f;
#pragma unroll
      for (int j = 0; j < 16; ++j) {
        int c = cq + j;
        float a = Asm[rr * 64 + c];
        float p = (a > 0.f) ? powf(a + 1e-12f, tau) : 0.f;
        so += a;
        ss += p;
        kf[rr * 64 + c] = (__bf16)p;
      }
      atomicAdd(&rs_o[rr], so);
      atomicAdd(&rs_s[rr], ss);
    }
#pragma unroll
    for (int it = 0; it < 16; ++it) {
      int q = tid + it * 256;
      int tt = q >> 6, c = q & 63;
      kb[c * 64 + tt] = (__bf16)vB[(size_t)(s0 + tt) * D_ + dbase + c];
    }
    __syncthreads();

    // phase 4: Y += A_sharp @ V
    {
      v16bf a0 = frag_ld(&kf[(mi * 16) * 64 + 0], 64, lane);
      v16bf a1 = frag_ld(&kf[(mi * 16) * 64 + 32], 64, lane);
#pragma unroll
      for (int j = 0; j < 2; ++j) {
        int n = nj + j;
        v16bf b0 = frag_ld(&kb[(n * 16) * 64 + 0], 64, lane);
        v16bf b1 = frag_ld(&kb[(n * 16) * 64 + 32], 64, lane);
        yacc[j] = wmma_bf16(a0, b0, yacc[j]);
        yacc[j] = wmma_bf16(a1, b1, yacc[j]);
      }
    }
    __syncthreads();
  }

  // per-row rescale rs_o/rs_s, dump Y tile to LDS (f32)
#pragma unroll
  for (int j = 0; j < 2; ++j) {
    int n = nj + j;
#pragma unroll
    for (int i = 0; i < 8; ++i) {
      int rloc = mi * 16 + lm + i;
      float o = fmaxf(rs_o[rloc], 1e-12f);
      float s = fmaxf(rs_s[rloc], 1e-12f);
      Asm[rloc * 64 + n * 16 + ln] = yacc[j][i] * (o / s);
    }
  }
  __syncthreads();

  // GroupNorm over K=64 per (t, head)
  if (tid < 64) {
    float mu = 0.f;
    for (int c = 0; c < 64; ++c) mu += Asm[tid * 64 + c];
    mu *= (1.f / 64.f);
    float var = 0.f;
    for (int c = 0; c < 64; ++c) {
      float dl = Asm[tid * 64 + c] - mu;
      var += dl * dl;
    }
    var *= (1.f / 64.f);
    cmf[tid] = mu;
    cmb[tid] = rsqrtf(var + 6.4e-4f);
  }
  __syncthreads();

#pragma unroll
  for (int it = 0; it < 16; ++it) {
    int q = tid + it * 256;
    int tt = q >> 6, c = q & 63;
    int t = t0 + tt, d = dbase + c;
    float val = (Asm[tt * 64 + c] - cmf[tt]) * cmb[tt];
    val = val * ln_g[d] + ln_b[d];
    val *= gB[(size_t)t * D_ + d];
    yg[(size_t)t * D_ + d] = (__bf16)val;
  }
}

// ---------------------------------------------------------------------------
extern "C" void kernel_launch(void* const* d_in, const int* in_sizes, int n_in,
                              void* d_out, int out_size, void* d_ws,
                              size_t ws_size, hipStream_t stream) {
  const float* x        = (const float*)d_in[0];
  const float* maa_x    = (const float*)d_in[1];
  const float* maa_w    = (const float*)d_in[2];
  const float* maa_kk   = (const float*)d_in[3];
  const float* maa_v    = (const float*)d_in[4];
  const float* maa_r    = (const float*)d_in[5];
  const float* maa_g    = (const float*)d_in[6];
  const float* maa_w1   = (const float*)d_in[7];
  const float* maa_w2   = (const float*)d_in[8];
  const float* time_dec = (const float*)d_in[9];
  const float* dw1      = (const float*)d_in[10];
  const float* dw2      = (const float*)d_in[11];
  const float* log_tau  = (const float*)d_in[12];
  const float* Wr       = (const float*)d_in[13];
  const float* Wk       = (const float*)d_in[14];
  const float* Wv       = (const float*)d_in[15];
  const float* Wg       = (const float*)d_in[16];
  const float* Wo       = (const float*)d_in[17];
  const float* ln_g     = (const float*)d_in[18];
  const float* ln_b     = (const float*)d_in[19];
  float* out = (float*)d_out;

  char* ws = (char*)d_ws;
  size_t o = 0;
  auto alloc = [&](size_t b) {
    size_t r = o;
    o += (b + 255) & ~(size_t)255;
    return r;
  };
  const size_t TD = (size_t)T_ * D_;
  __bf16* WrT    = (__bf16*)(ws + alloc((size_t)D_ * D_ * 2));
  __bf16* WkT    = (__bf16*)(ws + alloc((size_t)D_ * D_ * 2));
  __bf16* WvT    = (__bf16*)(ws + alloc((size_t)D_ * D_ * 2));
  __bf16* WgT    = (__bf16*)(ws + alloc((size_t)D_ * D_ * 2));
  __bf16* WoT    = (__bf16*)(ws + alloc((size_t)D_ * D_ * 2));
  __bf16* dw1T   = (__bf16*)(ws + alloc((size_t)128 * D_ * 2));  // 64->128 pad
  float*  xxx    = (float*)(ws + alloc((size_t)T_ * 160 * 4));
  __bf16* xw_bf  = (__bf16*)(ws + alloc(TD * 2));
  __bf16* xk_bf  = (__bf16*)(ws + alloc(TD * 2));
  __bf16* xv_bf  = (__bf16*)(ws + alloc(TD * 2));
  __bf16* xr_bf  = (__bf16*)(ws + alloc(TD * 2));
  __bf16* xg_bf  = (__bf16*)(ws + alloc(TD * 2));
  float*  rbuf   = (float*)(ws + alloc(TD * 4));
  float*  kbuf   = (float*)(ws + alloc(TD * 4));
  float*  vbuf   = (float*)(ws + alloc(TD * 4));
  float*  gbuf   = (float*)(ws + alloc(TD * 4));
  float*  h1     = (float*)(ws + alloc((size_t)T_ * 64 * 4));
  float*  whbuf  = (float*)(ws + alloc(TD * 4));
  float*  csbuf  = (float*)(ws + alloc(TD * 4));
  __bf16* yg_bf  = (__bf16*)(ws + alloc(TD * 2));

  const int nW = D_ * D_;
  cvt_transpose_k<<<(nW + 255) / 256, 256, 0, stream>>>(Wr, WrT, D_, D_, D_);
  cvt_transpose_k<<<(nW + 255) / 256, 256, 0, stream>>>(Wk, WkT, D_, D_, D_);
  cvt_transpose_k<<<(nW + 255) / 256, 256, 0, stream>>>(Wv, WvT, D_, D_, D_);
  cvt_transpose_k<<<(nW + 255) / 256, 256, 0, stream>>>(Wg, WgT, D_, D_, D_);
  cvt_transpose_k<<<(nW + 255) / 256, 256, 0, stream>>>(Wo, WoT, D_, D_, D_);
  cvt_transpose_k<<<(128 * D_ + 255) / 256, 256, 0, stream>>>(dw1, dw1T, 64, D_,
                                                              128);

  prep_k<<<T_, 256, 0, stream>>>(x, maa_x, maa_w1, xxx);
  mix_k<<<T_, 256, 0, stream>>>(x, xxx, maa_w2, maa_w, maa_kk, maa_v, maa_r,
                                maa_g, xw_bf, xk_bf, xv_bf, xr_bf, xg_bf);

  dim3 gFull(D_ / 128, T_ / 128);   // N=1024
  dim3 gDec(1, T_ / 128);           // N=64 (padded B rows)
  gemm_bf16_k<<<gFull, 256, 0, stream>>>(xr_bf, WrT, rbuf, D_, D_);
  gemm_bf16_k<<<gFull, 256, 0, stream>>>(xk_bf, WkT, kbuf, D_, D_);
  gemm_bf16_k<<<gFull, 256, 0, stream>>>(xv_bf, WvT, vbuf, D_, D_);
  gemm_bf16_k<<<gFull, 256, 0, stream>>>(xg_bf, WgT, gbuf, D_, D_);
  gemm_bf16_k<<<gDec, 256, 0, stream>>>(xw_bf, dw1T, h1, 64, D_);

  silu_k<<<(int)(TD / 256), 256, 0, stream>>>(gbuf, (int)TD);
  wdecay_k<<<T_, 256, 0, stream>>>(h1, dw2, time_dec, whbuf);
  cumsum_k<<<D_ / 256, 256, 0, stream>>>(whbuf, csbuf);

  attn_k<<<dim3(T_ / 64, H_), 256, 0, stream>>>(rbuf, kbuf, vbuf, csbuf, whbuf,
                                                log_tau, ln_g, ln_b, gbuf,
                                                yg_bf);

  gemm_bf16_k<<<gFull, 256, 0, stream>>>(yg_bf, WoT, out, D_, D_);
}
